// InfoNCELoss_42331197670081
// MI455X (gfx1250) — matrix operand order repaired and checked by reading, not compiled, not run on hardware
//
#include <hip/hip_runtime.h>
#include <hip/hip_bf16.h>

// ---------------------------------------------------------------------------
// Fused InfoNCE loss for MI455X (gfx1250, wave32, WMMA + async-LDS staging).
//   z  = normalize(concat(e0, e1))            (16384 x 128)
//   S  = z z^T / 0.07   (never materialized; flash-style streamed)
//   loss = mean_i( logsumexp_{j!=i} S_ij  -  S_{i, i^8192} )
// |S_ij| <= 1/0.07 -> FIXED-shift logsumexp, raw v_exp_f32 (arg in [-41.2,0]).
//
// Data movement: 8 waves/block share one column chunk; B tiles are staged
// once per block into LDS via GLOBAL_LOAD_ASYNC_TO_LDS_B128 (double-buffered
// 2 x 16KB), cutting L2 traffic 8x (4 GB -> 512 MB) vs per-wave global loads.
// ---------------------------------------------------------------------------

typedef __attribute__((ext_vector_type(16))) _Float16 v16h;
typedef __attribute__((ext_vector_type(4)))  _Float16 v4h;
typedef __attribute__((ext_vector_type(8)))  float    v8f;

#define B_ROWS   8192
#define N_TOT    16384
#define DIM      128
#define ROW_BYTES (DIM * 2)                   // 256 B per f16 row
#define INV_T    14.285714285714286f          // 1/0.07 == max |logit| == lse shift
#define LOG2E    1.4426950408889634f
#define K1       (INV_T * LOG2E)              // exp(l - INV_T) = exp2(c*K1 + K0)
#define K0       (-INV_T * LOG2E)
#define NEG_BIG  -1.0e9f
#define NCHUNK   4
#define TILES_N  (N_TOT / 16)                 // 1024 column tiles
#define TILES_PER_CHUNK (TILES_N / NCHUNK)    // 256
#define GROUP_TILES 4                         // column tiles staged per barrier
#define GROUP_ROWS  (GROUP_TILES * 16)        // 64 rows
#define GROUP_BYTES (GROUP_ROWS * ROW_BYTES)  // 16 KB
#define NGROUPS     (TILES_PER_CHUNK / GROUP_TILES)  // 64

// ---------------------------------------------------------------------------
// Kernel 1: one wave per row; f32 in -> L2-normalized f16 row out.
// ---------------------------------------------------------------------------
__global__ void __launch_bounds__(256)
infonce_normalize(const float* __restrict__ e0,
                  const float* __restrict__ e1,
                  _Float16* __restrict__ zh) {
    int row  = (blockIdx.x * blockDim.x + threadIdx.x) >> 5;
    int lane = threadIdx.x & 31;
    if (row >= N_TOT) return;
    const float* src = (row < B_ROWS) ? (e0 + (size_t)row * DIM)
                                      : (e1 + (size_t)(row - B_ROWS) * DIM);
    float4 v = reinterpret_cast<const float4*>(src)[lane];
    float ss = v.x * v.x + v.y * v.y + v.z * v.z + v.w * v.w;
#pragma unroll
    for (int m = 16; m >= 1; m >>= 1) ss += __shfl_xor(ss, m, 32);
    float scale = 1.0f / fmaxf(sqrtf(ss), 1e-12f);
    v4h h;
    h.x = (_Float16)(v.x * scale);
    h.y = (_Float16)(v.y * scale);
    h.z = (_Float16)(v.z * scale);
    h.w = (_Float16)(v.w * scale);
    reinterpret_cast<v4h*>(zh + (size_t)row * DIM)[lane] = h;
}

// ---------------------------------------------------------------------------
// B-operand registers (32x16 K x N per kc-step; ISA 7.12.2, slot->k matched
// to the A layout so both operands read untransposed rows of z):
//   lanes 0-15 : col n0+l,    halves z[kb..kb+7], z[kb+16..kb+23], kb = kc*32
//   lanes 16-31: col n0+l-16, halves z[kb+8..kb+15], z[kb+24..kb+31]
// ---------------------------------------------------------------------------
struct Bd { union { v16h h[4]; uint4 q[8]; }; };

// One 16x16 C tile: 4 chained WMMAs + streaming sum-of-exp epilogue.
// m0/n0 are SGPR-resident -> the special-tile test is a scalar branch; the
// fast path (1022/1024 tiles) is just v_pk_fma + v_exp_f32 + v_pk_add.
__device__ __forceinline__ void processTile(const v16h (&a)[4], const Bd& b,
                                            int n0, int m0, int half, int r15,
                                            float (&sAcc)[8], float (&pos)[8]) {
    v8f c = {};
#pragma unroll
    for (int kc = 0; kc < 4; ++kc)
        c = __builtin_amdgcn_wmma_f32_16x16x32_f16(
                false, a[kc], false, b.h[kc], (short)0, c, false, false);

    if (__builtin_expect((n0 == m0) || (n0 == (m0 ^ B_ROWS)), 0)) {
        int ncol = n0 + r15;
#pragma unroll
        for (int r = 0; r < 8; ++r) {
            int m   = m0 + r + half * 8;
            float l = c[r] * INV_T;
            pos[r]  = (ncol == (m ^ B_ROWS)) ? l : pos[r];
            float e = __builtin_amdgcn_exp2f((l - INV_T) * LOG2E);
            sAcc[r] += (ncol == m) ? 0.0f : e;            // mask self-sim
        }
    } else {
#pragma unroll
        for (int r = 0; r < 8; ++r)
            sAcc[r] += __builtin_amdgcn_exp2f(fmaf(c[r], K1, K0));
    }
}

// ---------------------------------------------------------------------------
// Kernel 2: block = 8 waves sharing one column chunk; wave w owns row strip
// m0 = (blockIdx>>2)*128 + w*16. B groups (64 rows = 16 KB) are staged
// global->LDS asynchronously, double buffered, and consumed via ds_load_b128.
// ---------------------------------------------------------------------------
__global__ void __launch_bounds__(256) __attribute__((amdgpu_waves_per_eu(1)))
infonce_simtile(const _Float16* __restrict__ zh,
                float* __restrict__ sPart,
                float* __restrict__ posPart) {
    __shared__ __align__(16) unsigned char sbuf[2][GROUP_BYTES];

    int tid   = threadIdx.x;
    int lane  = tid & 31;
    int w     = __builtin_amdgcn_readfirstlane(tid >> 5);  // wave in block, SGPR
    int chunk = blockIdx.x & (NCHUNK - 1);                 // uniform
    int m0    = ((int)(blockIdx.x >> 2) * 8 + w) * 16;     // uniform per wave
    int half  = lane >> 4;
    int r15   = lane & 15;
    const int colBase = chunk * (TILES_PER_CHUNK * 16);    // first column of chunk

    // A operand (16x32 per kc): lanes 0-15 row m0+l, halves z[kb..kb+7],
    // z[kb+16..kb+23]; lanes 16-31 row m0+l-16, halves z[kb+8..+15], z[kb+24..+31].
    const _Float16* aptr = zh + (size_t)(m0 + r15) * DIM;
    v16h a[4];
#pragma unroll
    for (int kc = 0; kc < 4; ++kc) {
        union { v16h h; uint4 q[2]; } u;
        int kb = kc * 32 + half * 8;
        u.q[0] = *reinterpret_cast<const uint4*>(aptr + kb);
        u.q[1] = *reinterpret_cast<const uint4*>(aptr + kb + 16);
        a[kc] = u.h;
    }

    float sAcc[8], pos[8];
#pragma unroll
    for (int r = 0; r < 8; ++r) { sAcc[r] = 0.0f; pos[r] = NEG_BIG; }

    // Stage group g (64 rows of z) into sbuf[bsel]: 1024 x 16B qwords split
    // across 256 threads x 4 async copies each (ASYNCcnt-tracked, no VGPRs).
    auto stage = [&](int g, int bsel) {
        const _Float16* src = zh + (size_t)(colBase + g * GROUP_ROWS) * DIM;
#pragma unroll
        for (int i = 0; i < 4; ++i) {
            int qi  = tid + 256 * i;            // 0..1023
            int row = qi >> 4, q = qi & 15;
            const void* gp = (const void*)(src + row * DIM + q * 8);
            unsigned lp = (unsigned)(size_t)(&sbuf[bsel][row * ROW_BYTES + q * 16]);
            asm volatile("global_load_async_to_lds_b128 %0, %1, off"
                         :: "v"(lp), "v"(gp) : "memory");
        }
    };

    // Compute the 4 column tiles of group g out of sbuf[bsel].
    auto computeGroup = [&](int g, int bsel) {
#pragma unroll
        for (int t = 0; t < GROUP_TILES; ++t) {
            int n0 = colBase + g * GROUP_ROWS + t * 16;
            const unsigned char* rowp =
                &sbuf[bsel][(t * 16 + r15) * ROW_BYTES + half * 16];
            Bd b;
#pragma unroll
            for (int kc = 0; kc < 4; ++kc) {
                b.q[2 * kc]     = *(const uint4*)(rowp + kc * 64);      // ds_load_b128
                b.q[2 * kc + 1] = *(const uint4*)(rowp + kc * 64 + 32);
            }
            processTile(a, b, n0, m0, half, r15, sAcc, pos);
        }
    };

    stage(0, 0);
    asm volatile("s_wait_asynccnt 0x0" ::: "memory");
    __syncthreads();
    for (int g = 0; g < NGROUPS; ++g) {
        int bsel = g & 1;
        if (g + 1 < NGROUPS) stage(g + 1, bsel ^ 1);   // overlap with compute
        computeGroup(g, bsel);
        asm volatile("s_wait_asynccnt 0x0" ::: "memory");  // own async done
        __syncthreads();   // all waves done with sbuf[bsel]; next buffer ready
    }

    // Row reduction across the 16 lanes holding each row (xor 1,2,4,8 keeps
    // lane halves 0-15 / 16-31 independent, matching rows 0-7 / 8-15).
#pragma unroll
    for (int m = 8; m >= 1; m >>= 1) {
#pragma unroll
        for (int r = 0; r < 8; ++r) {
            sAcc[r] += __shfl_xor(sAcc[r], m, 32);
            pos[r]  = fmaxf(pos[r], __shfl_xor(pos[r], m, 32));
        }
    }
    if (r15 == 0) {
#pragma unroll
        for (int r = 0; r < 8; ++r) {
            int m = m0 + r + half * 8;
            sPart  [(size_t)chunk * N_TOT + m] = sAcc[r];
            posPart[(size_t)chunk * N_TOT + m] = pos[r];
        }
    }
}

// ---------------------------------------------------------------------------
// Kernel 3: merge 4 chunk partials per row, loss_i = (log S_i + 1/T) - pos_i,
// mean over 16384 rows -> d_out[0]. Deterministic (no float atomics).
// ---------------------------------------------------------------------------
__global__ void __launch_bounds__(256)
infonce_finalize(const float* __restrict__ sPart,
                 const float* __restrict__ posPart,
                 float* __restrict__ out) {
    __shared__ float red[256];
    int t = threadIdx.x;
    float acc = 0.0f;
    for (int m = t; m < N_TOT; m += 256) {
        float S = 0.0f, p = NEG_BIG;
#pragma unroll
        for (int c = 0; c < NCHUNK; ++c) {
            S += sPart[(size_t)c * N_TOT + m];
            p  = fmaxf(p, posPart[(size_t)c * N_TOT + m]);
        }
        acc += (logf(S) + INV_T) - p;
    }
    red[t] = acc;
    __syncthreads();
    for (int s = 128; s > 0; s >>= 1) {
        if (t < s) red[t] += red[t + s];
        __syncthreads();
    }
    if (t == 0) out[0] = red[0] / (float)N_TOT;
}

// ---------------------------------------------------------------------------
extern "C" void kernel_launch(void* const* d_in, const int* in_sizes, int n_in,
                              void* d_out, int out_size, void* d_ws, size_t ws_size,
                              hipStream_t stream) {
    const float* e0 = (const float*)d_in[0];
    const float* e1 = (const float*)d_in[1];
    char* ws = (char*)d_ws;
    _Float16* zh   = (_Float16*)ws;                              // 4 MB (f16 z)
    float* sPart   = (float*)(ws + (size_t)N_TOT * DIM * 2);     // 256 KB
    float* posPart = sPart + (size_t)NCHUNK * N_TOT;             // 256 KB

    infonce_normalize<<<N_TOT / 8, 256, 0, stream>>>(e0, e1, zh);
    infonce_simtile<<<(TILES_N * NCHUNK) / 8, 256, 0, stream>>>(zh, sPart, posPart);
    infonce_finalize<<<1, 256, 0, stream>>>(sPart, posPart, (float*)d_out);
}
